// DuelingGNN_30640296689798
// MI455X (gfx1250) — matrix operand order repaired and checked by reading, not compiled, not run on hardware
//
#include <hip/hip_runtime.h>
#include <hip/hip_bf16.h>

typedef __attribute__((ext_vector_type(16))) _Float16 v16h;
typedef __attribute__((ext_vector_type(8)))  float    v8f;

#define LN_EPS 1e-5f

// ---------------- utility kernels ----------------
__global__ void k_zero(unsigned* p, int n) {
  int i = blockIdx.x * blockDim.x + threadIdx.x;
  int s = gridDim.x * blockDim.x;
  for (; i < n; i += s) p[i] = 0u;
}

__global__ void k_copy_i32(int* dst, const int* src, int n) {
  int i = blockIdx.x * blockDim.x + threadIdx.x;
  int s = gridDim.x * blockDim.x;
  for (; i < n; i += s) dst[i] = src[i];
}

__global__ void k_seed(int* mask, const int* cur) { mask[*cur] = 1; }

// One BFS hop: next[src] = 1 for every edge whose dst is in cur mask.
__global__ void k_hop(const int* __restrict__ esrc, const int* __restrict__ edst,
                      const int* __restrict__ cur, int* __restrict__ next, int E) {
  int e = blockIdx.x * blockDim.x + threadIdx.x;
  int s = gridDim.x * blockDim.x;
  for (; e < E; e += s)
    if (cur[edst[e]]) next[esrc[e]] = 1;
}

// deg = node_mask (self loop) ...
__global__ void k_deg_init(const int* __restrict__ mask, float* __restrict__ deg, int n) {
  int i = blockIdx.x * blockDim.x + threadIdx.x;
  int s = gridDim.x * blockDim.x;
  for (; i < n; i += s) deg[i] = mask[i] ? 1.f : 0.f;
}

// ... plus one per in-subgraph edge (adds of 1.0f are exact -> deterministic).
__global__ void k_deg_edges(const int* __restrict__ esrc, const int* __restrict__ edst,
                            const int* __restrict__ mask, float* __restrict__ deg, int E) {
  int e = blockIdx.x * blockDim.x + threadIdx.x;
  int s = gridDim.x * blockDim.x;
  for (; e < E; e += s) {
    int a = esrc[e], b = edst[e];
    if (mask[a] && mask[b]) atomicAdd(&deg[b], 1.f);
  }
}

// dis = mask ? rsqrt(deg) : 0   (in place)
__global__ void k_dis(const int* __restrict__ mask, float* __restrict__ degdis, int n) {
  int i = blockIdx.x * blockDim.x + threadIdx.x;
  int s = gridDim.x * blockDim.x;
  for (; i < n; i += s) degdis[i] = mask[i] ? rsqrtf(degdis[i]) : 0.f;
}

__global__ void k_compact_nodes(const int* __restrict__ mask, const float* __restrict__ dis,
                                int* __restrict__ node_list, int* __restrict__ local_id,
                                float* __restrict__ w_self, int* cnt, int n) {
  int i = blockIdx.x * blockDim.x + threadIdx.x;
  int s = gridDim.x * blockDim.x;
  for (; i < n; i += s)
    if (mask[i]) {
      int l = atomicAdd(cnt, 1);
      node_list[l] = i;
      local_id[i]  = l;
      w_self[l]    = dis[i] * dis[i];
    }
}

__global__ void k_compact_edges(const int* __restrict__ esrc, const int* __restrict__ edst,
                                const int* __restrict__ mask, const int* __restrict__ local_id,
                                const float* __restrict__ dis,
                                int* __restrict__ e_s, int* __restrict__ e_d,
                                float* __restrict__ e_w, int* ecnt, int E) {
  int e = blockIdx.x * blockDim.x + threadIdx.x;
  int s = gridDim.x * blockDim.x;
  for (; e < E; e += s) {
    int a = esrc[e], b = edst[e];
    if (mask[a] && mask[b]) {
      int j = atomicAdd(ecnt, 1);
      e_s[j] = local_id[a];
      e_d[j] = local_id[b];
      e_w[j] = dis[a] * dis[b];
    }
  }
}

__global__ void k_gather_x(const float* __restrict__ x, const int* __restrict__ node_list,
                           const int* cnt, float* __restrict__ h0, int din) {
  int count = *cnt;
  long long tot = (long long)count * din;
  long long i = blockIdx.x * blockDim.x + threadIdx.x;
  long long s = (long long)gridDim.x * blockDim.x;
  for (; i < tot; i += s) {
    int node = (int)(i / din), j = (int)(i - (long long)node * din);
    h0[i] = x[(size_t)node_list[node] * din + j];
  }
}

// ---------------- WMMA GEMM: out[count x DOUT] = h[count x DIN] @ W[DIN x DOUT] ----
// One wave per 16x16 output tile; K unrolled (DIN/32 wmma ops).
template <int DIN, int DOUT>
__global__ void k_gemm_wmma(const float* __restrict__ h, const float* __restrict__ W,
                            float* __restrict__ out, const int* cntPtr) {
  int count = *cntPtr;
  if (count <= 0) return;
  int tiles_m = (count + 15) >> 4;
  int tiles_n = DOUT >> 4;
  int total   = tiles_m * tiles_n;
  int lane = threadIdx.x & 31;
  int g = lane >> 4;            // half-wave group
  int r = lane & 15;
  int wave   = (blockIdx.x * blockDim.x + threadIdx.x) >> 5;
  int nwaves = (gridDim.x * blockDim.x) >> 5;
  for (int t = wave; t < total; t += nwaves) {
    int tm = t / tiles_n;
    int tn = t - tm * tiles_n;
    int row  = tm * 16 + r;     // A-matrix row owned by this lane
    int col0 = tn * 16;
    v8f c = {};
#pragma unroll
    for (int k0 = 0; k0 < DIN; k0 += 32) {
      // A 16x32 f16 layout: lane holds h[row][k0+8g .. +7] and h[row][k0+16+8g .. +7]
      v16h a;
      if (row < count) {
        const float* pa = h + (size_t)row * DIN + k0 + 8 * g;
#pragma unroll
        for (int j = 0; j < 8; ++j) a[j]     = (_Float16)pa[j];
#pragma unroll
        for (int j = 0; j < 8; ++j) a[8 + j] = (_Float16)pa[16 + j];
      } else {
#pragma unroll
        for (int j = 0; j < 16; ++j) a[j] = (_Float16)0.f;
      }
      // B 32x16 f16 layout: lane holds column col0+r, K = k0+8g .. +7 and k0+16+8g .. +7
      v16h b;
      const float* pb = W + (size_t)(k0 + 8 * g) * DOUT + col0 + r;
#pragma unroll
      for (int j = 0; j < 16; ++j) b[j] = (_Float16)pb[(size_t)(j < 8 ? j : 8 + j) * DOUT];
      c = __builtin_amdgcn_wmma_f32_16x16x32_f16(false, a, false, b, (short)0, c,
                                                 false, false);
    }
    // C/D f32 layout: lane -> N = col0+r ; VGPR p -> M = tm*16 + p + 8g
#pragma unroll
    for (int p = 0; p < 8; ++p) {
      int m = tm * 16 + p + 8 * g;
      if (m < count) out[(size_t)m * DOUT + col0 + r] = c[p];
    }
  }
}

// agg = hW * w_self + bias
__global__ void k_agg_init(const float* __restrict__ hW, const float* __restrict__ w_self,
                           const float* __restrict__ bias, float* __restrict__ agg,
                           const int* cnt, int dout) {
  int count = *cnt;
  long long tot = (long long)count * dout;
  long long i = blockIdx.x * blockDim.x + threadIdx.x;
  long long s = (long long)gridDim.x * blockDim.x;
  for (; i < tot; i += s) {
    int node = (int)(i / dout), j = (int)(i - (long long)node * dout);
    agg[i] = hW[i] * w_self[node] + bias[j];
  }
}

// agg[dst] += w * hW[src]
__global__ void k_agg_edges(const float* __restrict__ hW, const int* __restrict__ e_s,
                            const int* __restrict__ e_d, const float* __restrict__ e_w,
                            float* __restrict__ agg, const int* ecnt, int dout) {
  int ec = *ecnt;
  long long tot = (long long)ec * dout;
  long long i = blockIdx.x * blockDim.x + threadIdx.x;
  long long s = (long long)gridDim.x * blockDim.x;
  for (; i < tot; i += s) {
    int e = (int)(i / dout), j = (int)(i - (long long)e * dout);
    atomicAdd(&agg[(size_t)e_d[e] * dout + j], e_w[e] * hW[(size_t)e_s[e] * dout + j]);
  }
}

// ReLU + LayerNorm, one wave32 per node (lane holds DOUT/32 features)
template <int DOUT>
__global__ void k_relu_ln(const float* __restrict__ agg, const float* __restrict__ gamma,
                          const float* __restrict__ beta, float* __restrict__ hout,
                          const int* cnt) {
  constexpr int PER = DOUT / 32;
  int count = *cnt;
  int lane = threadIdx.x & 31;
  int wave   = (blockIdx.x * blockDim.x + threadIdx.x) >> 5;
  int nwaves = (gridDim.x * blockDim.x) >> 5;
  for (int i = wave; i < count; i += nwaves) {
    float v[PER];
    float s = 0.f;
#pragma unroll
    for (int j = 0; j < PER; ++j) {
      float t = agg[(size_t)i * DOUT + lane * PER + j];
      t = fmaxf(t, 0.f);
      v[j] = t;
      s += t;
    }
#pragma unroll
    for (int m = 16; m >= 1; m >>= 1) s += __shfl_xor(s, m, 32);
    float mu = s * (1.f / DOUT);
    float q = 0.f;
#pragma unroll
    for (int j = 0; j < PER; ++j) { float d = v[j] - mu; q += d * d; }
#pragma unroll
    for (int m = 16; m >= 1; m >>= 1) q += __shfl_xor(q, m, 32);
    float inv = rsqrtf(q * (1.f / DOUT) + LN_EPS);
#pragma unroll
    for (int j = 0; j < PER; ++j) {
      int col = lane * PER + j;
      hout[(size_t)i * DOUT + col] = (v[j] - mu) * inv * gamma[col] + beta[col];
    }
  }
}

// mean pool over subgraph nodes
__global__ void k_pool(const float* __restrict__ h, const int* cnt, float* __restrict__ out,
                       int dout) {
  int count = *cnt;
  float invc = count > 0 ? 1.f / (float)count : 0.f;
  long long tot = (long long)count * dout;
  long long i = blockIdx.x * blockDim.x + threadIdx.x;
  long long s = (long long)gridDim.x * blockDim.x;
  for (; i < tot; i += s) {
    int node = (int)(i / dout), j = (int)(i - (long long)node * dout);
    atomicAdd(&out[j], h[(size_t)node * dout + j] * invc);
  }
}

// ---------------- host launch ----------------
extern "C" void kernel_launch(void* const* d_in, const int* in_sizes, int n_in,
                              void* d_out, int out_size, void* d_ws, size_t ws_size,
                              hipStream_t stream) {
  (void)n_in; (void)ws_size;
  const float* x   = (const float*)d_in[0];
  const int*   ei  = (const int*)d_in[1];
  const int*   cur = (const int*)d_in[2];
  const float* W1  = (const float*)d_in[3];
  const float* b1  = (const float*)d_in[4];
  const float* g1  = (const float*)d_in[5];
  const float* be1 = (const float*)d_in[6];
  const float* W2  = (const float*)d_in[7];
  const float* b2  = (const float*)d_in[8];
  const float* g2  = (const float*)d_in[9];
  const float* be2 = (const float*)d_in[10];
  const float* W3  = (const float*)d_in[11];
  const float* b3  = (const float*)d_in[12];
  const float* g3  = (const float*)d_in[13];
  const float* be3 = (const float*)d_in[14];

  const int DIN = 128, DH = 128, DOUT = 64;
  const int N = in_sizes[0] / DIN;
  const int E = in_sizes[1] / 2;
  const int* esrc = ei;
  const int* edst = ei + E;

  // workspace carve (256B aligned)
  char* wsp = (char*)d_ws;
  size_t off = 0;
  auto take = [&](size_t bytes) -> void* {
    void* p = wsp + off;
    off = (off + bytes + 255) & ~(size_t)255;
    return p;
  };
  int*   cnt       = (int*)  take(2 * sizeof(int)); // [0]=node count, [1]=edge count
  int*   mask_a    = (int*)  take((size_t)N * 4);
  int*   mask_b    = (int*)  take((size_t)N * 4);
  int*   local_id  = (int*)  take((size_t)N * 4);
  int*   node_list = (int*)  take((size_t)N * 4);
  float* degdis    = (float*)take((size_t)N * 4);
  float* w_self    = (float*)take((size_t)N * 4);
  int*   e_s       = (int*)  take((size_t)E * 4);
  int*   e_d       = (int*)  take((size_t)E * 4);
  float* e_w       = (float*)take((size_t)E * 4);
  float* H0        = (float*)take((size_t)N * DH * 4);
  float* H1        = (float*)take((size_t)N * DH * 4);

  const int T = 256;
  int gN = (N + T - 1) / T;
  int gE = (E + T - 1) / T;

  // init
  k_zero<<<gN, T, 0, stream>>>((unsigned*)mask_a, N);
  k_zero<<<gN, T, 0, stream>>>((unsigned*)mask_b, N);
  k_zero<<<1, 64, 0, stream>>>((unsigned*)cnt, 2);
  k_zero<<<1, 64, 0, stream>>>((unsigned*)d_out, out_size);
  k_seed<<<1, 1, 0, stream>>>(mask_a, cur);

  // 2-hop BFS (double buffered, exactly 2 hops like the reference)
  k_copy_i32<<<gN, T, 0, stream>>>(mask_b, mask_a, N);
  k_hop<<<gE, T, 0, stream>>>(esrc, edst, mask_a, mask_b, E);
  k_copy_i32<<<gN, T, 0, stream>>>(mask_a, mask_b, N);
  k_hop<<<gE, T, 0, stream>>>(esrc, edst, mask_b, mask_a, E);

  // degrees -> dis, compact nodes and edges to local index space
  k_deg_init<<<gN, T, 0, stream>>>(mask_a, degdis, N);
  k_deg_edges<<<gE, T, 0, stream>>>(esrc, edst, mask_a, degdis, E);
  k_dis<<<gN, T, 0, stream>>>(mask_a, degdis, N);
  k_compact_nodes<<<gN, T, 0, stream>>>(mask_a, degdis, node_list, local_id, w_self, cnt, N);
  k_compact_edges<<<gE, T, 0, stream>>>(esrc, edst, mask_a, local_id, degdis,
                                        e_s, e_d, e_w, cnt + 1, E);

  // gather features of subgraph nodes
  k_gather_x<<<2048, T, 0, stream>>>(x, node_list, cnt, H0, DIN);

  // layer 1: H0 -> gemm -> H1 -> agg -> H0 -> relu+LN -> H1
  k_gemm_wmma<128, 128><<<256, T, 0, stream>>>(H0, W1, H1, cnt);
  k_agg_init<<<2048, T, 0, stream>>>(H1, w_self, b1, H0, cnt, DH);
  k_agg_edges<<<2048, T, 0, stream>>>(H1, e_s, e_d, e_w, H0, cnt + 1, DH);
  k_relu_ln<128><<<256, T, 0, stream>>>(H0, g1, be1, H1, cnt);

  // layer 2: H1 -> H0 -> H1 -> H0
  k_gemm_wmma<128, 128><<<256, T, 0, stream>>>(H1, W2, H0, cnt);
  k_agg_init<<<2048, T, 0, stream>>>(H0, w_self, b2, H1, cnt, DH);
  k_agg_edges<<<2048, T, 0, stream>>>(H0, e_s, e_d, e_w, H1, cnt + 1, DH);
  k_relu_ln<128><<<256, T, 0, stream>>>(H1, g2, be2, H0, cnt);

  // layer 3 (dout=64): H0 -> H1 -> H0 -> H1
  k_gemm_wmma<128, 64><<<256, T, 0, stream>>>(H0, W3, H1, cnt);
  k_agg_init<<<2048, T, 0, stream>>>(H1, w_self, b3, H0, cnt, DOUT);
  k_agg_edges<<<2048, T, 0, stream>>>(H1, e_s, e_d, e_w, H0, cnt + 1, DOUT);
  k_relu_ln<64><<<256, T, 0, stream>>>(H0, g3, be3, H1, cnt);

  // mean pool over subgraph
  k_pool<<<512, T, 0, stream>>>(H1, cnt, (float*)d_out, DOUT);
}
